// DeepGraphConvLayer_17300128269006
// MI455X (gfx1250) — matrix operand order repaired and checked by reading, not compiled, not run on hardware
//
#include <hip/hip_runtime.h>
#include <hip/hip_bf16.h>

typedef __attribute__((ext_vector_type(16))) _Float16 v16h;
typedef __attribute__((ext_vector_type(8)))  float    v8f;
typedef __attribute__((__vector_size__(16))) int      v4i_vec;   // matches builtin param type

#define IN_FEATS 256
#define HEADS 4
#define HEAD_DIM 64
#define NEG_SLOPE 0.2f
#define BN_EPS 1e-5f

#if __has_builtin(__builtin_amdgcn_global_load_async_to_lds_b128) && \
    __has_builtin(__builtin_amdgcn_s_wait_asynccnt)
#define HAS_ASYNC_LDS 1
#else
#define HAS_ASYNC_LDS 0
#endif

// ---------- helpers: order-preserving float<->uint map for atomic max ----------
__device__ __forceinline__ unsigned orderedFromFloat(float f) {
  unsigned u = __float_as_uint(f);
  return (u & 0x80000000u) ? ~u : (u | 0x80000000u);
}
__device__ __forceinline__ float floatFromOrdered(unsigned o) {
  unsigned u = (o & 0x80000000u) ? (o & 0x7FFFFFFFu) : ~o;
  return __uint_as_float(u);
}

// ---------- 0) zero init of accumulator region ----------
__global__ void zero_init_kernel(float* p, size_t n) {
  size_t i = (size_t)blockIdx.x * blockDim.x + threadIdx.x;
  if (i < n) p[i] = 0.0f;
}

// ---------- 1) pack W_src/W_dst (f32, [K=256][N=256]) into f16 WMMA B layout ----------
// Flat packed index: which*65536 + kt*8192 + nt*512 + lane*16 + j
// B-operand layout (16-bit, 32x16): lanes 0-15 hold K=0..7,16..23 of col N=lane&15,
// lanes 16-31 hold K=8..15,24..31 of col N=lane&15.
__global__ void pack_weights_kernel(const float* Wsrc, const float* Wdst, _Float16* Wp) {
  int idx = blockIdx.x * blockDim.x + threadIdx.x;   // 0 .. 131071
  if (idx >= 2 * 8 * 16 * 512) return;
  int j     = idx & 15;
  int lane  = (idx >> 4) & 31;
  int nt    = (idx >> 9) & 15;
  int kt    = (idx >> 13) & 7;
  int which = idx >> 16;
  const float* W = which ? Wdst : Wsrc;
  int K = kt * 32 + (j & 7) + ((lane < 16) ? 0 : 8) + ((j >> 3) << 4);
  int N = nt * 16 + (lane & 15);
  Wp[idx] = (_Float16)W[K * IN_FEATS + N];
}

// ---------- 2) projection GEMM via v_wmma_f32_16x16x32_f16 ----------
// 256-thread block = 8 waves = 4 M-tiles x 2 N-halves. Each wave: 16x128 slice,
// 8 v8f accumulators. B K-slab (16KB) staged in LDS, double-buffered via
// global_load_async_to_lds_b128 so the copy overlaps the WMMA chain.
__global__ void __launch_bounds__(256)
gemm_proj_kernel(const float* __restrict__ feats,
                 const _Float16* __restrict__ Wp,
                 const float* __restrict__ b_src,
                 const float* __restrict__ b_dst,
                 float* __restrict__ fs, float* __restrict__ fd,
                 int N_nodes, int mtiles) {
#if HAS_ASYNC_LDS
  __shared__ _Float16 ldsB[2][8192];         // 32 KB double buffer
#else
  __shared__ _Float16 ldsB[1][8192];         // 16 KB single buffer
#endif
  int t    = threadIdx.x;
  int wave = t >> 5;                         // 0..7
  int lane = t & 31;
  int mtile = blockIdx.x * 4 + (wave >> 1);
  if (mtile >= mtiles) mtile = mtiles - 1;   // clamp, keep barrier convergence
  int nhalf = wave & 1;                      // which 8 N-tiles this wave owns
  int which = blockIdx.y;                    // 0 = src proj, 1 = dst proj
  const _Float16* Wbase = Wp + (size_t)which * 65536;
  const float* bias = which ? b_dst : b_src;
  float* out = which ? fd : fs;

  int m0 = mtile * 16;
  int arow = m0 + (lane & 15);
  if (arow >= N_nodes) arow = N_nodes - 1;
  int kbase = (lane < 16) ? 0 : 8;           // A layout: lanes 16-31 hold K+8 / K+24

#if HAS_ASYNC_LDS
  auto stage_async = [&](int buf, int kt) {
    const _Float16* g = Wbase + (size_t)kt * 8192;
#pragma unroll
    for (int i = 0; i < 4; ++i) {
      int idx = i * 256 + t;                 // 16-byte chunk index
      __builtin_amdgcn_global_load_async_to_lds_b128(
          (__attribute__((address_space(1))) v4i_vec*)(g + idx * 8),
          (__attribute__((address_space(3))) v4i_vec*)(&ldsB[buf][idx * 8]), 0, 0);
    }
  };
  stage_async(0, 0);                         // prologue: slab 0 in flight
#endif

  v8f acc[8];
#pragma unroll
  for (int nt = 0; nt < 8; ++nt) acc[nt] = (v8f){0.f,0.f,0.f,0.f,0.f,0.f,0.f,0.f};

  for (int kt = 0; kt < 8; ++kt) {
#if HAS_ASYNC_LDS
    int cur = kt & 1;
    __builtin_amdgcn_s_wait_asynccnt(0);     // slab kt landed in LDS
    __syncthreads();                         // ...for every wave; prior reads done
    if (kt < 7) stage_async(cur ^ 1, kt + 1);  // overlap next copy with compute
#else
    const int cur = 0;
    __syncthreads();                         // previous iteration's LDS reads done
    {
      const _Float16* g = Wbase + (size_t)kt * 8192;
#pragma unroll
      for (int i = 0; i < 4; ++i) {
        int idx = i * 256 + t;
        ((uint4*)&ldsB[0][0])[idx] = ((const uint4*)g)[idx];
      }
    }
    __syncthreads();                         // staged B visible to all waves
#endif

    // ---- A operand (16-bit, 16x32): halves 0-7 = K+[0..7], halves 8-15 = K+16+[0..7]
    const float* ap = feats + (size_t)arow * IN_FEATS + kt * 32 + kbase;
    __builtin_prefetch(ap + 32, 0, 1);       // global_prefetch_b8
    v16h a;
#pragma unroll
    for (int j = 0; j < 8; ++j) a[j] = (_Float16)ap[j];
#pragma unroll
    for (int j = 0; j < 8; ++j) a[8 + j] = (_Float16)ap[16 + j];

    const _Float16* lb = &ldsB[cur][0] + (size_t)(nhalf * 8) * 512 + lane * 16;
    // preload B tiles in batches of 4 so WMMAs issue back-to-back instead of
    // serializing on each ds_load pair
#pragma unroll
    for (int g4 = 0; g4 < 2; ++g4) {
      v16h bt[4];
#pragma unroll
      for (int q = 0; q < 4; ++q) bt[q] = *(const v16h*)(lb + (g4 * 4 + q) * 512);
#pragma unroll
      for (int q = 0; q < 4; ++q) {
        acc[g4 * 4 + q] = __builtin_amdgcn_wmma_f32_16x16x32_f16(
            /*neg_a=*/false, a, /*neg_b=*/false, bt[q],
            /*c_mod=*/(short)0, acc[g4 * 4 + q], /*reuse_a=*/false, /*reuse_b=*/false);
      }
    }
  }

  // C/D layout: VGPR r -> row r (lanes 0-15) or row 8+r (lanes 16-31), col = lane&15
  int ncol = lane & 15;
  int mofs = (lane < 16) ? 0 : 8;
  if (m0 + 16 <= N_nodes) {                  // full tile: unguarded stores
#pragma unroll
    for (int nt = 0; nt < 8; ++nt) {
      int n = (nhalf * 8 + nt) * 16 + ncol;
      float bv = bias[n];
      float* op = out + (size_t)(m0 + mofs) * IN_FEATS + n;
#pragma unroll
      for (int r = 0; r < 8; ++r) op[(size_t)r * IN_FEATS] = acc[nt][r] + bv;
    }
  } else {                                   // tail tile: guarded stores
#pragma unroll
    for (int nt = 0; nt < 8; ++nt) {
      int n = (nhalf * 8 + nt) * 16 + ncol;
      float bv = bias[n];
#pragma unroll
      for (int r = 0; r < 8; ++r) {
        int row = m0 + mofs + r;
        if (row < N_nodes) out[(size_t)row * IN_FEATS + n] = acc[nt][r] + bv;
      }
    }
  }
}

// ---------- 3) per-edge attention logits + segment max ----------
// One wave per edge; lane l handles flat channels [8l, 8l+8); head = l>>3.
__global__ void edge_logits_kernel(const float* __restrict__ fs,
                                   const float* __restrict__ fd,
                                   const int* __restrict__ src,
                                   const int* __restrict__ dst,
                                   const float* __restrict__ attn_w,
                                   float* __restrict__ elog,
                                   unsigned* __restrict__ mmax, int E) {
  int e = blockIdx.x * 8 + (threadIdx.x >> 5);
  if (e >= E) return;
  int lane = threadIdx.x & 31;
  int s = src[e], d = dst[e];
  const float* ps = fs + (size_t)s * IN_FEATS + lane * 8;
  const float* pd = fd + (size_t)d * IN_FEATS + lane * 8;
  const float* pa = attn_w + lane * 8;
  float acc = 0.f;
#pragma unroll
  for (int j = 0; j < 8; ++j) {
    float v = ps[j] + pd[j];
    v = (v > 0.f) ? v : NEG_SLOPE * v;       // LeakyReLU
    acc += v * pa[j];
  }
  // reduce across the 8 lanes of each head (wave32 shuffles)
  acc += __shfl_xor(acc, 1, 32);
  acc += __shfl_xor(acc, 2, 32);
  acc += __shfl_xor(acc, 4, 32);
  if ((lane & 7) == 0) {
    int h = lane >> 3;
    elog[(size_t)e * HEADS + h] = acc;
    atomicMax(&mmax[(size_t)d * HEADS + h], orderedFromFloat(acc));
  }
}

// ---------- 4) exp(logit - max) + segment sum ----------
__global__ void edge_exp_kernel(float* __restrict__ elog,
                                const int* __restrict__ dst,
                                const unsigned* __restrict__ mmax,
                                float* __restrict__ ssum, int E) {
  int idx = blockIdx.x * blockDim.x + threadIdx.x;
  if (idx >= E * HEADS) return;
  int e = idx >> 2, h = idx & 3;
  int d = dst[e];
  float m = floatFromOrdered(mmax[(size_t)d * HEADS + h]);
  float ex = __expf(elog[idx] - m);
  elog[idx] = ex;
  atomicAdd(&ssum[(size_t)d * HEADS + h], ex);
}

// ---------- 5) normalize attention + weighted scatter-add aggregation ----------
__global__ void edge_aggregate_kernel(const float* __restrict__ fs,
                                      const float* __restrict__ elog,
                                      const float* __restrict__ ssum,
                                      const int* __restrict__ src,
                                      const int* __restrict__ dst,
                                      float* __restrict__ rst,
                                      float* __restrict__ attn_out, int E) {
  int e = blockIdx.x * 8 + (threadIdx.x >> 5);
  if (e >= E) return;
  int lane = threadIdx.x & 31;
  int h = lane >> 3;
  int s = src[e], d = dst[e];
  float ex = elog[(size_t)e * HEADS + h];
  float a = ex / ssum[(size_t)d * HEADS + h];
  if ((lane & 7) == 0) attn_out[(size_t)e * HEADS + h] = a;
  const float* ps = fs + (size_t)s * IN_FEATS + lane * 8;
  float* pr = rst + (size_t)d * IN_FEATS + lane * 8;
#pragma unroll
  for (int j = 0; j < 8; ++j) atomicAdd(&pr[j], ps[j] * a);
}

// ---------- 6) residual + bias, write pre-BN output, block-reduced channel stats ----------
__global__ void bn_pre_stats_kernel(const float* __restrict__ rst,
                                    const float* __restrict__ feats,
                                    const float* __restrict__ bias,
                                    float* __restrict__ out,
                                    float* __restrict__ sums, int N_nodes) {
  int c = threadIdx.x;                       // 256 threads = 256 channels
  int r0 = blockIdx.x * 128;
  int r1 = min(r0 + 128, N_nodes);
  float bv = bias[c];
  float s = 0.f, s2 = 0.f;
  for (int r = r0; r < r1; ++r) {
    size_t i = (size_t)r * IN_FEATS + c;
    float x = rst[i] + feats[i] + bv;
    out[i] = x;
    s += x; s2 += x * x;
  }
  atomicAdd(&sums[c], s);
  atomicAdd(&sums[IN_FEATS + c], s2);
}

// ---------- 7) BatchNorm (batch stats, biased var) + ReLU in place ----------
__global__ void bn_norm_kernel(float* __restrict__ out,
                               const float* __restrict__ sums,
                               const float* __restrict__ gamma,
                               const float* __restrict__ beta, int N_nodes) {
  size_t idx = (size_t)blockIdx.x * blockDim.x + threadIdx.x;
  if (idx >= (size_t)N_nodes * IN_FEATS) return;
  int c = idx & (IN_FEATS - 1);
  float invN = 1.0f / (float)N_nodes;
  float mean = sums[c] * invN;
  float var = sums[IN_FEATS + c] * invN - mean * mean;
  float x = out[idx];
  float y = (x - mean) * rsqrtf(var + BN_EPS) * gamma[c] + beta[c];
  out[idx] = fmaxf(y, 0.0f);
}

extern "C" void kernel_launch(void* const* d_in, const int* in_sizes, int n_in,
                              void* d_out, int out_size, void* d_ws, size_t ws_size,
                              hipStream_t stream) {
  const float* features = (const float*)d_in[0];
  const int*   src      = (const int*)d_in[1];
  const int*   dst      = (const int*)d_in[2];
  const float* W_src    = (const float*)d_in[3];
  const float* b_src    = (const float*)d_in[4];
  const float* W_dst    = (const float*)d_in[5];
  const float* b_dst    = (const float*)d_in[6];
  const float* attn_w   = (const float*)d_in[7];
  const float* bias     = (const float*)d_in[8];
  const float* gamma    = (const float*)d_in[9];
  const float* beta     = (const float*)d_in[10];

  int N = in_sizes[0] / IN_FEATS;      // 50000
  int E = in_sizes[1];                 // 800000

  float* out      = (float*)d_out;                         // [N,256]
  float* attn_out = out + (size_t)N * IN_FEATS;            // [E,4]

  // workspace layout (floats); zero-init region first, all chunks 32B aligned
  float* ws = (float*)d_ws;
  size_t o = 0;
  float*    rst  = ws + o; o += (size_t)N * IN_FEATS;      // aggregation accumulator
  float*    ssum = ws + o; o += (size_t)N * HEADS;         // segment sums
  unsigned* mmax = (unsigned*)(ws + o); o += (size_t)N * HEADS;  // segment max (ordered uint)
  float*    sums = ws + o; o += 2 * IN_FEATS;              // BN channel sums
  size_t zcount = o;
  float*    elog = ws + o; o += (size_t)E * HEADS;         // logits -> exp
  float*    fs   = ws + o; o += (size_t)N * IN_FEATS;      // src projection
  float*    fd   = ws + o; o += (size_t)N * IN_FEATS;      // dst projection
  _Float16* Wp   = (_Float16*)(ws + o);                    // 2*64K packed f16 weights

  int mtiles = (N + 15) / 16;

  zero_init_kernel<<<(int)((zcount + 255) / 256), 256, 0, stream>>>(ws, zcount);
  pack_weights_kernel<<<(2 * 8 * 16 * 512) / 256, 256, 0, stream>>>(W_src, W_dst, Wp);
  gemm_proj_kernel<<<dim3((mtiles + 3) / 4, 2), 256, 0, stream>>>(
      features, Wp, b_src, b_dst, fs, fd, N, mtiles);
  edge_logits_kernel<<<(E + 7) / 8, 256, 0, stream>>>(fs, fd, src, dst, attn_w, elog, mmax, E);
  edge_exp_kernel<<<(E * HEADS + 255) / 256, 256, 0, stream>>>(elog, dst, mmax, ssum, E);
  edge_aggregate_kernel<<<(E + 7) / 8, 256, 0, stream>>>(fs, elog, ssum, src, dst, rst, attn_out, E);
  bn_pre_stats_kernel<<<(N + 127) / 128, 256, 0, stream>>>(rst, features, bias, out, sums, N);
  bn_norm_kernel<<<(int)(((size_t)N * IN_FEATS + 255) / 256), 256, 0, stream>>>(out, sums, gamma, beta, N);
}
